// SinkhornDistanceLoss_83313775607971
// MI455X (gfx1250) — matrix operand order repaired and checked by reading, not compiled, not run on hardware
//
#include <hip/hip_runtime.h>

// ---------------------------------------------------------------------------
// Sinkhorn loss (geomloss-style eps-scaling) for sequence (8,9,128,32,32).
// Fused flash-attention-style softmin sweeps: the 1024x1024 cost matrix is
// never materialized; x.y dot tiles are rebuilt every sweep with
// v_wmma_f32_16x16x32_f16 from an L2-resident 19MB f16 feature buffer.
// y-tiles are streamed into double-buffered LDS with CDNA5 async
// global->LDS DMA (global_load_async_to_lds_b128 / s_wait_asynccnt).
// Online logsumexp runs in the log2 domain (v_exp_f32 is natively base-2).
// ---------------------------------------------------------------------------

typedef __attribute__((ext_vector_type(16))) _Float16 v16h;
typedef __attribute__((ext_vector_type(8)))  float    v8f;

#define NBATCH   8
#define NTIME    9
#define NCH      128
#define NPTS     1024
#define NFRAMES  (NBATCH * NTIME)      // 72
#define NPAIR    (NBATCH * (NTIME-1))  // 64
#define NITER    30
#define HB       (-6.93147180559945f)  // -log(1024), uniform weights
#define LOG2E    1.44269504088896341f
#define LN2      0.69314718055994531f
#define YSTRIDE  136                   // padded LDS row stride in halves (16B aligned, bank-spread)

// ---------------------------------------------------------------------------
// Kernel 1: fp32 [frame][ch][pt] -> f16 [frame][pt][ch] + squared norms.
// ---------------------------------------------------------------------------
__global__ void prep_kernel(const float* __restrict__ seq,
                            _Float16* __restrict__ feat,
                            float* __restrict__ norms) {
  const int frame = blockIdx.x >> 2;
  const int pt    = ((blockIdx.x & 3) << 8) + threadIdx.x;
  const float* src = seq + (size_t)frame * NCH * NPTS + pt;   // stride NPTS per channel
  _Float16*    dst = feat + ((size_t)frame * NPTS + pt) * NCH;
  float nrm = 0.f;
  for (int cb = 0; cb < 16; ++cb) {
    union { uint4 u; _Float16 h[8]; } pk;
    #pragma unroll
    for (int k = 0; k < 8; ++k) {
      float v = src[(size_t)(cb * 8 + k) * NPTS];
      nrm += v * v;
      pk.h[k] = (_Float16)v;
    }
    *reinterpret_cast<uint4*>(dst + cb * 8) = pk.u;           // 16B aligned
  }
  norms[(size_t)frame * NPTS + pt] = nrm;
}

// ---------------------------------------------------------------------------
// Kernel 2: per-channel max/min over all 72*1024 points.
// ---------------------------------------------------------------------------
__global__ void chminmax_kernel(const float* __restrict__ seq,
                                float* __restrict__ chmax,
                                float* __restrict__ chmin) {
  __shared__ float smax[256], smin[256];
  const int ch = blockIdx.x;
  float mx = -3.4e38f, mn = 3.4e38f;
  for (int idx = threadIdx.x; idx < NFRAMES * NPTS; idx += 256) {
    int fr = idx >> 10, n = idx & 1023;
    float v = seq[((size_t)fr * NCH + ch) * NPTS + n];
    mx = fmaxf(mx, v);
    mn = fminf(mn, v);
  }
  smax[threadIdx.x] = mx; smin[threadIdx.x] = mn;
  __syncthreads();
  for (int s = 128; s > 0; s >>= 1) {
    if ((int)threadIdx.x < s) {
      smax[threadIdx.x] = fmaxf(smax[threadIdx.x], smax[threadIdx.x + s]);
      smin[threadIdx.x] = fminf(smin[threadIdx.x], smin[threadIdx.x + s]);
    }
    __syncthreads();
  }
  if (threadIdx.x == 0) { chmax[ch] = smax[0]; chmin[ch] = smin[0]; }
}

// ---------------------------------------------------------------------------
// Kernel 3: diameter = ||max - min||_2 + 1e-6
// ---------------------------------------------------------------------------
__global__ void diam_kernel(const float* __restrict__ chmax,
                            const float* __restrict__ chmin,
                            float* __restrict__ diam) {
  __shared__ float ss[128];
  float d = chmax[threadIdx.x] - chmin[threadIdx.x];
  ss[threadIdx.x] = d * d;
  __syncthreads();
  for (int s = 64; s > 0; s >>= 1) {
    if ((int)threadIdx.x < s) ss[threadIdx.x] += ss[threadIdx.x + s];
    __syncthreads();
  }
  if (threadIdx.x == 0) *diam = sqrtf(ss[0]) + 1e-6f;
}

// ---------------------------------------------------------------------------
// Kernel 4: fused softmin sweep.
//   out[p,i] = -eps * LSE_j( HB + (h_j - C_ij)/eps ),  C_ij = max(0, .5(|x|^2+|y|^2) - x.y)
// optionally folded: out = 0.5*(foldsrc + value).
// dir=0: rows = frame t, cols = frame t+1 (f-update); dir=1: swapped (g-update).
// Block: 256 threads = 8 waves; wave owns 16 rows; 128-row tile per block.
// ---------------------------------------------------------------------------
__launch_bounds__(256)
__global__ void softmin_kernel(const _Float16* __restrict__ feat,
                               const float* __restrict__ norms,
                               const float* __restrict__ hsrc,     // g (or f) raw, may be null
                               const float* __restrict__ foldsrc,  // old f (or g), may be null
                               float* __restrict__ outv,
                               const float* __restrict__ diam_p,
                               int t_iter, int dir) {
  __shared__ _Float16 ys[2][128 * YSTRIDE];   // double-buffered y tile (async DMA target)
  __shared__ float yn_s[2][128];
  __shared__ float hc_s[2][128];

  const int p       = blockIdx.y;
  const int rowtile = blockIdx.x;
  const int bb = p >> 3, ti = p & 7;
  const int rframe = bb * NTIME + ti + dir;        // row (query) frame
  const int cframe = bb * NTIME + ti + 1 - dir;    // col (key) frame

  // eps from annealing ladder (device-side: depends on diameter)
  float eps;
  if (t_iter < 0) {
    eps = 0.01f;                                    // blur^2
  } else {
    float diam = *diam_p;
    float frac = (float)t_iter * (1.0f / 29.0f);
    float l = (1.0f - frac) * log2f(diam) + frac * (-3.32192809488736f); // log2(0.1)
    eps = exp2f(2.0f * l);                          // (diam*(0.1/diam)^frac)^2
  }
  // log2-domain scaling: score2 = log2(e)*(HB + (h_j - C_ij)/eps)
  const float inv_eps2 = LOG2E / eps;
  const float hb2      = HB * LOG2E;

  const int tid  = threadIdx.x;
  const int wave = tid >> 5;
  const int lane = tid & 31;
  const int l15  = lane & 15;
  const int h8   = lane >> 4;
  const int growbase = rowtile * 128 + wave * 16;

  // --- hoist A fragments (this wave's 16 rows, K = 0..127) into registers ---
  // A 16x32 f16 layout: lane = row; halves = {K=kb*32+h8*8+0..7, K=kb*32+16+h8*8+0..7}
  union Frag { uint4 u[2]; v16h v; };
  Frag a[4];
  {
    const uint4* arow = reinterpret_cast<const uint4*>(
        feat + ((size_t)rframe * NPTS + growbase + l15) * NCH);
    #pragma unroll
    for (int kb = 0; kb < 4; ++kb) {
      a[kb].u[0] = arow[kb * 4 + h8];
      a[kb].u[1] = arow[kb * 4 + 2 + h8];
    }
  }

  // per-lane row state: C/D layout => VGPR v holds row (8*h8 + v)
  float xn[8], m[8], s[8];
  #pragma unroll
  for (int v = 0; v < 8; ++v) {
    xn[v] = norms[(size_t)rframe * NPTS + growbase + 8 * h8 + v];
    m[v] = -1e30f;
    s[v] = 0.0f;
  }

  const float* hrow = hsrc ? (hsrc + (size_t)p * NPTS) : nullptr;

  // -------- async y-tile issue: 8 x b128 per thread, LDS written by DMA -----
  auto issue_tile = [&](int j0, int buf) {
    const unsigned gbase = (unsigned)(((size_t)cframe * NPTS + j0) * (NCH * 2));
    #pragma unroll
    for (int it = 0; it < 8; ++it) {
      int idx = it * 256 + tid;
      int ptl = idx >> 4, ck = idx & 15;
      unsigned goff = gbase + (unsigned)(ptl * (YSTRIDE * 0 + 256) + ck * 16);
      unsigned ldsa = (unsigned)(uintptr_t)(&ys[buf][ptl * YSTRIDE + ck * 8]);
      asm volatile("global_load_async_to_lds_b128 %0, %1, %2"
                   :: "v"(ldsa), "v"(goff), "s"(feat)
                   : "memory");
    }
    if (tid < 128) {
      yn_s[buf][tid] = norms[(size_t)cframe * NPTS + j0 + tid];
      hc_s[buf][tid] = hrow ? hrow[j0 + tid] : 0.0f;
    }
  };

  issue_tile(0, 0);

  for (int jt = 0; jt < 8; ++jt) {
    const int buf = jt & 1;
    if (jt < 7) {
      issue_tile((jt + 1) * 128, buf ^ 1);
      // in-order async completion: <=8 outstanding => this tile's 8 are done
      asm volatile("s_wait_asynccnt 0x8" ::: "memory");
    } else {
      asm volatile("s_wait_asynccnt 0x0" ::: "memory");
    }
    __syncthreads();   // tile jt (all waves' DMA) now visible

    #pragma unroll
    for (int cb = 0; cb < 8; ++cb) {
      v8f acc = {0.f, 0.f, 0.f, 0.f, 0.f, 0.f, 0.f, 0.f};
      const int brow = cb * 16 + l15;   // column point this lane serves
      #pragma unroll
      for (int kb = 0; kb < 4; ++kb) {
        // B 32x16 f16 layout: lane = col; halves = {K=kb*32+h8*8+0..7, K=kb*32+16+h8*8+0..7}
        Frag b;
        const uint4* bp =
            reinterpret_cast<const uint4*>(&ys[buf][brow * YSTRIDE + kb * 32 + h8 * 8]);
        b.u[0] = bp[0];
        b.u[1] = bp[2];
        acc = __builtin_amdgcn_wmma_f32_16x16x32_f16(
            false, a[kb].v, false, b.v, (short)0, acc, false, false);
      }
      const float ynj = yn_s[buf][cb * 16 + l15];
      const float hj  = hc_s[buf][cb * 16 + l15];
      #pragma unroll
      for (int v = 0; v < 8; ++v) {
        float dot = acc[v];
        float Cij = fmaxf(0.0f, 0.5f * (xn[v] + ynj) - dot);
        float sc  = hb2 + (hj - Cij) * inv_eps2;       // log2-domain score
        float M   = fmaxf(m[v], sc);
        s[v] = s[v] * exp2f(m[v] - M) + exp2f(sc - M); // native v_exp_f32
        m[v] = M;
      }
    }
    __syncthreads();   // all waves done reading buf before its DMA reuse
  }

  // merge the 16 lanes that share each output row (online-LSE combine, log2 dom)
  #pragma unroll
  for (int v = 0; v < 8; ++v) {
    float mv = m[v], sv = s[v];
    #pragma unroll
    for (int off = 1; off < 16; off <<= 1) {
      float m2 = __shfl_xor(mv, off, 32);
      float s2 = __shfl_xor(sv, off, 32);
      float M  = fmaxf(mv, m2);
      sv = sv * exp2f(mv - M) + s2 * exp2f(m2 - M);
      mv = M;
    }
    m[v] = mv; s[v] = sv;
  }

  if (l15 == 0) {
    #pragma unroll
    for (int v = 0; v < 8; ++v) {
      int row = growbase + 8 * h8 + v;
      float val = -eps * LN2 * (m[v] + __log2f(s[v]));   // back to natural log
      if (foldsrc) val = 0.5f * (foldsrc[(size_t)p * NPTS + row] + val);
      outv[(size_t)p * NPTS + row] = val;
    }
  }
}

// ---------------------------------------------------------------------------
// Kernel 5: loss = (sum f_fin + sum g_fin) / (P*N); f_fin,g_fin contiguous.
// ---------------------------------------------------------------------------
__global__ void reduce_kernel(const float* __restrict__ fin, float* __restrict__ out) {
  __shared__ float ss[256];
  float acc = 0.f;
  for (int i = threadIdx.x; i < 2 * NPAIR * NPTS; i += 256) acc += fin[i];
  ss[threadIdx.x] = acc;
  __syncthreads();
  for (int s = 128; s > 0; s >>= 1) {
    if ((int)threadIdx.x < s) ss[threadIdx.x] += ss[threadIdx.x + s];
    __syncthreads();
  }
  if (threadIdx.x == 0) out[0] = ss[0] * (1.0f / (float)(NPAIR * NPTS));
}

// ---------------------------------------------------------------------------
extern "C" void kernel_launch(void* const* d_in, const int* in_sizes, int n_in,
                              void* d_out, int out_size, void* d_ws, size_t ws_size,
                              hipStream_t stream) {
  (void)in_sizes; (void)n_in; (void)out_size; (void)ws_size;
  const float* seq = (const float*)d_in[0];

  // workspace carve-out (~20.3 MB)
  char* ws = (char*)d_ws;
  size_t off = 0;
  _Float16* feat  = (_Float16*)(ws + off); off += (size_t)NFRAMES * NPTS * NCH * 2;
  float* norms    = (float*)(ws + off);    off += (size_t)NFRAMES * NPTS * 4;
  float* f0       = (float*)(ws + off);    off += (size_t)NPAIR * NPTS * 4;
  float* g0       = (float*)(ws + off);    off += (size_t)NPAIR * NPTS * 4;
  float* f1       = (float*)(ws + off);    off += (size_t)NPAIR * NPTS * 4;  // f1,g1 contiguous
  float* g1       = (float*)(ws + off);    off += (size_t)NPAIR * NPTS * 4;
  float* chmax    = (float*)(ws + off);    off += 128 * 4;
  float* chmin    = (float*)(ws + off);    off += 128 * 4;
  float* diam     = (float*)(ws + off);    off += 256;

  prep_kernel<<<dim3(NFRAMES * 4), 256, 0, stream>>>(seq, feat, norms);
  chminmax_kernel<<<dim3(128), 256, 0, stream>>>(seq, chmax, chmin);
  diam_kernel<<<dim3(1), 128, 0, stream>>>(chmax, chmin, diam);

  const dim3 sg(8, NPAIR);  // 8 row tiles x 64 pairs

  // init: f = softmin(eps0, C, hb); g = softmin(eps0, C^T, hb)
  softmin_kernel<<<sg, 256, 0, stream>>>(feat, norms, nullptr, nullptr, f0, diam, 0, 0);
  softmin_kernel<<<sg, 256, 0, stream>>>(feat, norms, nullptr, nullptr, g0, diam, 0, 1);

  // 30 annealed iterations; parallel update needs old (f,g) -> double buffer
  float* fc = f0; float* gc = g0; float* fn = f1; float* gn = g1;
  for (int t = 0; t < NITER; ++t) {
    softmin_kernel<<<sg, 256, 0, stream>>>(feat, norms, gc, fc, fn, diam, t, 0);
    softmin_kernel<<<sg, 256, 0, stream>>>(feat, norms, fc, gc, gn, diam, t, 1);
    float* tf = fc; fc = fn; fn = tf;
    float* tg = gc; gc = gn; gn = tg;
  }

  // final extrapolation at eps = blur^2 = 0.01 (no folding)
  softmin_kernel<<<sg, 256, 0, stream>>>(feat, norms, gc, nullptr, fn, diam, -1, 0);
  softmin_kernel<<<sg, 256, 0, stream>>>(feat, norms, fc, nullptr, gn, diam, -1, 1);

  // fn/gn are an adjacent (f,g) buffer pair -> single contiguous reduction
  reduce_kernel<<<1, 256, 0, stream>>>(fn, (float*)d_out);
}
